// MultiHeadAttention_45440753992004
// MI455X (gfx1250) — compile-verified, matrix operand order checked by reading, not run on hardware
//
#include <hip/hip_runtime.h>

// ---------------------------------------------------------------------------
// MI455X (gfx1250) fused MHA: bf16 WMMA for projections/QK^T/PV, f32 WMMA for
// the streaming FC. wave32 everywhere; lane = tid & 31.
// ---------------------------------------------------------------------------

typedef __attribute__((ext_vector_type(16))) __bf16 v16bf;
typedef __attribute__((ext_vector_type(8)))  __bf16 v8bf;
typedef __attribute__((ext_vector_type(8)))  float  v8f;
typedef __attribute__((ext_vector_type(2)))  float  v2f;

#define WMMA_BF16(a, b, c) \
  __builtin_amdgcn_wmma_f32_16x16x32_bf16(false, (a), false, (b), (short)0, (c), false, false)
#define WMMA_F32(a, b, c) \
  __builtin_amdgcn_wmma_f32_16x16x4_f32(false, (a), false, (b), (short)0, (c), false, false)

#define CAT16(lo, hi) \
  __builtin_shufflevector((lo), (hi), 0, 1, 2, 3, 4, 5, 6, 7, 8, 9, 10, 11, 12, 13, 14, 15)

constexpr int Bc    = 16;
constexpr int Lc    = 512;
constexpr int Dc    = 512;
constexpr int Hc    = 8;
constexpr int DKc   = 64;
constexpr int DVc   = 64;
constexpr int HDc   = Hc * DKc;          // 512
constexpr int FCIN  = Lc * Hc * DVc;     // 262144
constexpr int FCOUT = HDc;               // 512
constexpr int NSPLIT = 128;
constexpr int KSPL   = FCIN / NSPLIT;    // 2048

// Pack 16 f32 (two contiguous runs of 8) into a bf16 A/B fragment.
// Element order: elems 0..7 = K {kb+0..7}, elems 8..15 = K {kb+16..23}
// (per ISA 7.12.2: 16-bit A 16x32, lanes 0-15 hold K 0..7/16..23,
//  lanes 16-31 hold K 8..15/24..31 -> caller applies kb = laneHi ? 8 : 0).
static __device__ inline v16bf pack4(float4 a, float4 b, float4 c, float4 d) {
  v16bf r;
  r[0]  = (__bf16)a.x; r[1]  = (__bf16)a.y; r[2]  = (__bf16)a.z; r[3]  = (__bf16)a.w;
  r[4]  = (__bf16)b.x; r[5]  = (__bf16)b.y; r[6]  = (__bf16)b.z; r[7]  = (__bf16)b.w;
  r[8]  = (__bf16)c.x; r[9]  = (__bf16)c.y; r[10] = (__bf16)c.z; r[11] = (__bf16)c.w;
  r[12] = (__bf16)d.x; r[13] = (__bf16)d.y; r[14] = (__bf16)d.z; r[15] = (__bf16)d.w;
  return r;
}

// ---------------------------------------------------------------------------
// Projection GEMM: Y[M,N] = X[M,K] * W[N,K]^T + bias,  M=8192 N=512 K=512.
// mode 0: Y row-major bf16 [B*L, H*dk]
// mode 1: Y transposed per head: [H, B, DV, L] bf16   (for the V operand of PV)
// Block = 256 threads (8 waves); each wave owns one 16x16 C tile.
// ---------------------------------------------------------------------------
__global__ __launch_bounds__(256) void proj_gemm(
    const float* __restrict__ X, const float* __restrict__ W,
    const float* __restrict__ bias, __bf16* __restrict__ Y, int mode) {
  const int lane   = threadIdx.x & 31;
  const int wave   = threadIdx.x >> 5;
  const int laneHi = lane >> 4;
  const int lr     = lane & 15;
  const int kb     = laneHi ? 8 : 0;

  const int mtile = blockIdx.x * 16;
  const int ntile = blockIdx.y * 128 + wave * 16;

  const float* xr = X + (size_t)(mtile + lr) * Dc;
  const float* wr = W + (size_t)(ntile + lr) * Dc;

  v8f c = {};
  for (int k0 = 0; k0 < Dc; k0 += 32) {
    const float4* xa = (const float4*)(xr + k0 + kb);
    const float4* xb = (const float4*)(xr + k0 + kb + 16);
    v16bf amat = pack4(xa[0], xa[1], xb[0], xb[1]);
    const float4* wa = (const float4*)(wr + k0 + kb);
    const float4* wb = (const float4*)(wr + k0 + kb + 16);
    v16bf bmat = pack4(wa[0], wa[1], wb[0], wb[1]);
    c = WMMA_BF16(amat, bmat, c);
  }

  const int gcol = ntile + lr;
  const float bb = bias[gcol];
  const int r0   = mtile + (laneHi ? 8 : 0);
  if (mode == 0) {
    for (int r = 0; r < 8; ++r)
      Y[(size_t)(r0 + r) * HDc + gcol] = (__bf16)(c[r] + bb);
  } else {
    const int hh = gcol >> 6, dd = gcol & 63;
    for (int r = 0; r < 8; ++r) {
      const int grow = r0 + r;
      const int b2 = grow >> 9, ll = grow & 511;
      Y[(((size_t)hh * Bc + b2) * DVc + dd) * Lc + ll] = (__bf16)(c[r] + bb);
    }
  }
}

// ---------------------------------------------------------------------------
// Attention: one block (4 waves, 128 thr) per (h, b, 16-row q tile).
//   Phase A: S = Q K^T / 8 via bf16 WMMA -> 32KB LDS tile [16][512]
//   Phase B: row softmax over 512 keys, write attn (f32, d_out)
//   Phase C: O = P V via bf16 WMMA (P repacked from LDS, V pre-transposed)
// ---------------------------------------------------------------------------
__global__ __launch_bounds__(128) void attn_kernel(
    const __bf16* __restrict__ qw, const __bf16* __restrict__ kw,
    const __bf16* __restrict__ vt, float* __restrict__ attn,
    float* __restrict__ o) {
  __shared__ float s[16][512];
  __shared__ float red[16][8];
  __shared__ float rowstat[16];

  const int lane   = threadIdx.x & 31;
  const int wave   = threadIdx.x >> 5;
  const int laneHi = lane >> 4;
  const int lr     = lane & 15;
  const int kb     = laneHi ? 8 : 0;

  const int qtile = blockIdx.x * 16;
  const int b     = blockIdx.y;
  const int h     = blockIdx.z;

  // ---- Phase A: scores ----
  const __bf16* qp = qw + ((size_t)(b * Lc + qtile + lr) * Hc + h) * DKc;
  v16bf aq0 = CAT16(*(const v8bf*)(qp + kb),       *(const v8bf*)(qp + kb + 16));
  v16bf aq1 = CAT16(*(const v8bf*)(qp + 32 + kb),  *(const v8bf*)(qp + 32 + kb + 16));

  for (int i = 0; i < 8; ++i) {
    const int nt   = wave * 8 + i;               // 32 N-tiles over 4 waves
    const __bf16* kp = kw + ((size_t)(b * Lc + nt * 16 + lr) * Hc + h) * DKc;
    v16bf bk0 = CAT16(*(const v8bf*)(kp + kb),      *(const v8bf*)(kp + kb + 16));
    v16bf bk1 = CAT16(*(const v8bf*)(kp + 32 + kb), *(const v8bf*)(kp + 32 + kb + 16));
    v8f c = {};
    c = WMMA_BF16(aq0, bk0, c);
    c = WMMA_BF16(aq1, bk1, c);
    const int mr = laneHi ? 8 : 0;
    for (int r = 0; r < 8; ++r)
      s[mr + r][nt * 16 + lr] = c[r] * 0.125f;   // 1/sqrt(64)
  }
  __syncthreads();

  // ---- Phase B: softmax (8 threads per row) ----
  const int row = threadIdx.x >> 3;
  const int sub = threadIdx.x & 7;
  float m = -3.402823466e+38f;
  for (int cidx = sub; cidx < 512; cidx += 8) m = fmaxf(m, s[row][cidx]);
  red[row][sub] = m;
  __syncthreads();
  if (sub == 0) {
    float mm = red[row][0];
    for (int j = 1; j < 8; ++j) mm = fmaxf(mm, red[row][j]);
    rowstat[row] = mm;
  }
  __syncthreads();
  const float mm = rowstat[row];
  float sum = 0.0f;
  for (int cidx = sub; cidx < 512; cidx += 8) {
    float e = __expf(s[row][cidx] - mm);
    s[row][cidx] = e;
    sum += e;
  }
  red[row][sub] = sum;
  __syncthreads();
  if (sub == 0) {
    float t = red[row][0];
    for (int j = 1; j < 8; ++j) t += red[row][j];
    rowstat[row] = t;
  }
  __syncthreads();
  const float inv = 1.0f / rowstat[row];
  float* arow = attn + ((size_t)(h * Bc + b) * Lc + qtile + row) * Lc;
  for (int cidx = sub; cidx < 512; cidx += 8) {
    float p = s[row][cidx] * inv;
    s[row][cidx] = p;
    arow[cidx]   = p;
  }
  __syncthreads();

  // ---- Phase C: O = P * V  (each wave: one 16-col tile of DV) ----
  const int dcol = wave * 16 + lr;
  const __bf16* vrow = vt + (((size_t)h * Bc + b) * DVc + dcol) * Lc;
  v8f c2 = {};
  for (int k0 = 0; k0 < 512; k0 += 32) {
    const float4* pa = (const float4*)&s[lr][k0 + kb];
    const float4* pb = (const float4*)&s[lr][k0 + kb + 16];
    v16bf ap = pack4(pa[0], pa[1], pb[0], pb[1]);
    v16bf bp = CAT16(*(const v8bf*)(vrow + k0 + kb), *(const v8bf*)(vrow + k0 + kb + 16));
    c2 = WMMA_BF16(ap, bp, c2);
  }
  const int mr = laneHi ? 8 : 0;
  for (int r = 0; r < 8; ++r)
    o[((size_t)(b * Lc + qtile + mr + r) * Hc + h) * DVc + dcol] = c2[r];
}

// ---------------------------------------------------------------------------
// Final FC: out[16,512] = A[16,262144] * Wf[512,262144]^T + bf.
// Memory-bound on the 512MB Wf stream: f32 WMMA (16x16x4), NT loads +
// prefetch so Wf doesn't thrash L2. Split-K=128 partials -> deterministic
// reduce kernel (no float atomics).
// ---------------------------------------------------------------------------
__global__ __launch_bounds__(32) void fc_partial(
    const float* __restrict__ A, const float* __restrict__ Wf,
    float* __restrict__ partial) {
  const int lane   = threadIdx.x & 31;
  const int laneHi = lane >> 4;
  const int lr     = lane & 15;
  const int ntile  = blockIdx.x * 16;
  const int split  = blockIdx.y;
  const int ko     = laneHi ? 2 : 0;   // f32 A 16x4: lanes 0-15 K={0,1}, 16-31 K={2,3}

  const float* ar = A  + (size_t)lr * FCIN          + (size_t)split * KSPL + ko;
  const float* wr = Wf + (size_t)(ntile + lr) * FCIN + (size_t)split * KSPL + ko;

  v8f c = {};
  for (int k = 0; k < KSPL; k += 4) {
    __builtin_prefetch(wr + k + 512, 0, 1);
    v2f av = { ar[k], ar[k + 1] };
    v2f wv = { __builtin_nontemporal_load(wr + k),
               __builtin_nontemporal_load(wr + k + 1) };
    c = WMMA_F32(av, wv, c);
  }

  const int mr = laneHi ? 8 : 0;
  float* pp = partial + (size_t)split * (16 * FCOUT);
  for (int r = 0; r < 8; ++r)
    pp[(mr + r) * FCOUT + ntile + lr] = c[r];
}

__global__ __launch_bounds__(256) void fc_reduce(
    const float* __restrict__ partial, const float* __restrict__ bfv,
    float* __restrict__ out) {
  const int i = blockIdx.x * 256 + threadIdx.x;    // 0..8191
  float acc = bfv[i & (FCOUT - 1)];
  for (int sidx = 0; sidx < NSPLIT; ++sidx)
    acc += partial[(size_t)sidx * (16 * FCOUT) + i];
  out[i] = acc;
}

// ---------------------------------------------------------------------------
// Launch. Inputs: q,k,v,Wq,bq,Wk,bk,Wv,bv,Wf,bf.  Output: [out(8192) | attn].
// Workspace: qw(8MB bf16) kw(8MB) vt(8MB) o(16MB f32) partial(4MB) = 44MB.
// ---------------------------------------------------------------------------
extern "C" void kernel_launch(void* const* d_in, const int* in_sizes, int n_in,
                              void* d_out, int out_size, void* d_ws, size_t ws_size,
                              hipStream_t stream) {
  const float* q  = (const float*)d_in[0];
  const float* k  = (const float*)d_in[1];
  const float* v  = (const float*)d_in[2];
  const float* Wq = (const float*)d_in[3];
  const float* bq = (const float*)d_in[4];
  const float* Wk = (const float*)d_in[5];
  const float* bk = (const float*)d_in[6];
  const float* Wv = (const float*)d_in[7];
  const float* bv = (const float*)d_in[8];
  const float* Wf = (const float*)d_in[9];
  const float* bf = (const float*)d_in[10];

  char* ws = (char*)d_ws;
  __bf16* qw      = (__bf16*)(ws);
  __bf16* kw      = (__bf16*)(ws + 8388608ull);
  __bf16* vt      = (__bf16*)(ws + 2ull * 8388608ull);
  float*  o       = (float*)(ws + 3ull * 8388608ull);
  float*  partial = (float*)(ws + 3ull * 8388608ull + 16777216ull);

  float* out  = (float*)d_out;
  float* attn = out + Bc * FCOUT;   // output is 16*512 = 8192 floats, attn follows

  dim3 gp(Bc * Lc / 16, HDc / 128);           // (512, 4)
  proj_gemm<<<gp, 256, 0, stream>>>(q, Wq, bq, qw, 0);
  proj_gemm<<<gp, 256, 0, stream>>>(k, Wk, bk, kw, 0);
  proj_gemm<<<gp, 256, 0, stream>>>(v, Wv, bv, vt, 1);

  dim3 ga(Lc / 16, Bc, Hc);                   // (32, 16, 8)
  attn_kernel<<<ga, 128, 0, stream>>>(qw, kw, vt, attn, o);

  dim3 gf(FCOUT / 16, NSPLIT);                // (32, 128)
  fc_partial<<<gf, 32, 0, stream>>>(o, Wf, partial);
  fc_reduce<<<Bc * FCOUT / 256, 256, 0, stream>>>(partial, bf, out);
}
